// SResTransformerTrain_42228118454951
// MI455X (gfx1250) — compile-verified, matrix operand order checked by reading, not run on hardware
//
#include <hip/hip_runtime.h>
#include <hip/hip_bf16.h>
#include <math.h>

// ---------------- model constants ----------------
#define BATCH   2
#define SEQ     2048
#define DMODEL  1024
#define NHEAD   16
#define DHEAD   64
#define NLAYER  8
#define DFF     4096
#define MROWS   (BATCH * SEQ)       // 4096
#define GRIDW   64
#define CHUNK   16

typedef __attribute__((ext_vector_type(16))) _Float16 v16h;
typedef __attribute__((ext_vector_type(8)))  _Float16 v8h;
typedef __attribute__((ext_vector_type(8)))  float    v8f;
typedef __attribute__((ext_vector_type(2)))  float    v2f;
typedef __attribute__((ext_vector_type(4)))  unsigned v4u;
typedef __attribute__((ext_vector_type(8)))  int      v8i;
typedef __attribute__((ext_vector_type(4)))  int      v4i;

// Use the CDNA5 Tensor Data Mover when the toolchain exposes it.
#if defined(__HIP_DEVICE_COMPILE__) && defined(__has_builtin)
#if __has_builtin(__builtin_amdgcn_tensor_load_to_lds) && __has_builtin(__builtin_amdgcn_s_wait_tensorcnt)
#define USE_TDM 1
#endif
#endif
#ifndef USE_TDM
#define USE_TDM 0
#endif

// ---------------- GEMM tile config ----------------
#define BM 128
#define BN 64
#define BK 32
#define LDH 40         // halves per LDS tile row: 32 data + 4-DWORD pad -> 80B (16B aligned)
#define A_TILE_H 5120  // 128*40 halves = 10240 B
#define B_TILE_H 2560  // 64*40 halves  = 5120 B

#if USE_TDM
// Issue one 2D TDM load: tile (tile_rows x 32 halves) from row-major f16 [rows,K]
// into LDS at lds_byte_addr with 40-half padded row stride.
__device__ __forceinline__ void tdm_issue_2d(unsigned lds_byte_addr,
                                             const _Float16* gptr,
                                             int K, int tensor_rows, int tile_rows)
{
    unsigned long long ga = (unsigned long long)(uintptr_t)gptr;
    v4u g0;
    g0[0] = 1u;                                          // count=1, user mode
    g0[1] = lds_byte_addr;                               // lds_addr
    g0[2] = (unsigned)(ga & 0xFFFFFFFFull);              // global_addr[31:0]
    g0[3] = (unsigned)((ga >> 32) & 0x1FFFFFFull) | (2u << 30);  // addr[56:32] | type=2
    v8i g1;
    // data_size=1 (2B), pad_enable, pad_interval=3 (16 DW), pad_amount=3 (4 DW)
    g1[0] = (int)((1u << 16) | (1u << 20) | (3u << 22) | (3u << 25));
    g1[1] = (int)(((unsigned)K & 0xFFFFu) << 16);                         // tensor_dim0 lo16
    g1[2] = (int)((((unsigned)K >> 16) & 0xFFFFu) |
                  (((unsigned)tensor_rows & 0xFFFFu) << 16));             // dim0 hi | dim1 lo
    g1[3] = (int)((((unsigned)tensor_rows >> 16) & 0xFFFFu) | (32u << 16)); // dim1 hi | tile_dim0=32
    g1[4] = (int)((unsigned)tile_rows & 0xFFFFu);                         // tile_dim1 | tile_dim2=0
    g1[5] = (int)(unsigned)K;                                             // tensor_dim0_stride lo32
    g1[6] = 0;
    g1[7] = 0;
    v4i g2 = {0, 0, 0, 0};
    v4i g3 = {0, 0, 0, 0};
    v8i g4 = {0, 0, 0, 0, 0, 0, 0, 0};
    __builtin_amdgcn_tensor_load_to_lds(g0, g1, g2, g3, g4, 0);
}
#endif

// C[M,N] = A16[M,K] @ Wt16[N,K]^T (+bias) (+relu) (+residual)
template<bool RELU, bool HASRES, bool OUTH>
__global__ __launch_bounds__(256)
void gemm_wmma(const _Float16* __restrict__ A, const _Float16* __restrict__ Bt,
               const float* __restrict__ bias, const float* __restrict__ Res,
               float* __restrict__ C32, _Float16* __restrict__ C16,
               int M, int N, int K)
{
    // double-buffered tiles: A0 A1 B0 B1  (30720 B total)
    __shared__ __align__(16) _Float16 smem[2 * A_TILE_H + 2 * B_TILE_H];

    const int tid  = threadIdx.x;
    const int lane = tid & 31;
    const int wave = tid >> 5;
    const int wm   = wave >> 1;        // 0..3
    const int wn   = wave & 1;         // 0..1
    const int bm0  = blockIdx.y * BM;
    const int bn0  = blockIdx.x * BN;
    const int l15  = lane & 15;
    const int khalf = (lane < 16) ? 0 : 8;

    v8f acc[2][2] = {};
    const int ntiles = K / BK;

#if USE_TDM
    const unsigned lds_base = (unsigned)(uintptr_t)(&smem[0]);
    if (wave == 0) {
        tdm_issue_2d(lds_base,                  A + (size_t)bm0 * K, K, M, BM);
        tdm_issue_2d(lds_base + 4u * A_TILE_H,  Bt + (size_t)bn0 * K, K, N, BN);
    }
#endif

    for (int t = 0; t < ntiles; ++t) {
        const int buf = t & 1;
        _Float16* As = smem + buf * A_TILE_H;
        _Float16* Bs = smem + 2 * A_TILE_H + buf * B_TILE_H;

#if USE_TDM
        if (wave == 0) {
            if (t + 1 < ntiles) {
                const int nb = (t + 1) & 1;
                const int k1 = (t + 1) * BK;
                tdm_issue_2d(lds_base + (unsigned)(nb * 2 * A_TILE_H),
                             A + (size_t)bm0 * K + k1, K, M, BM);
                tdm_issue_2d(lds_base + (unsigned)(4 * A_TILE_H + nb * 2 * B_TILE_H),
                             Bt + (size_t)bn0 * K + k1, K, N, BN);
                __builtin_amdgcn_s_wait_tensorcnt(2);   // oldest stage landed
            } else {
                __builtin_amdgcn_s_wait_tensorcnt(0);
            }
        }
        __syncthreads();
#else
        {   // synchronous fallback staging (f16 vector copies)
            const int k0 = t * BK;
            int r  = tid >> 1;
            int hb = (tid & 1) * 16;
            v8h a0 = *(const v8h*)(A + (size_t)(bm0 + r) * K + k0 + hb);
            v8h a1 = *(const v8h*)(A + (size_t)(bm0 + r) * K + k0 + hb + 8);
            *(v8h*)&As[r * LDH + hb]     = a0;
            *(v8h*)&As[r * LDH + hb + 8] = a1;
            if (tid < 128) {
                v8h b0 = *(const v8h*)(Bt + (size_t)(bn0 + r) * K + k0 + hb);
                v8h b1 = *(const v8h*)(Bt + (size_t)(bn0 + r) * K + k0 + hb + 8);
                *(v8h*)&Bs[r * LDH + hb]     = b0;
                *(v8h*)&Bs[r * LDH + hb + 8] = b1;
            }
            __syncthreads();
        }
#endif

        // ---- fragments: two aligned 16B LDS reads each ----
        v16h afrag[2], bfrag[2];
        #pragma unroll
        for (int mi = 0; mi < 2; ++mi) {
            int mrow = wm * 32 + mi * 16 + l15;
            v8h lo = *(const v8h*)&As[mrow * LDH + khalf];
            v8h hi = *(const v8h*)&As[mrow * LDH + khalf + 16];
            afrag[mi] = __builtin_shufflevector(lo, hi, 0,1,2,3,4,5,6,7,8,9,10,11,12,13,14,15);
        }
        #pragma unroll
        for (int ni = 0; ni < 2; ++ni) {
            int nrow = wn * 32 + ni * 16 + l15;
            v8h lo = *(const v8h*)&Bs[nrow * LDH + khalf];
            v8h hi = *(const v8h*)&Bs[nrow * LDH + khalf + 16];
            bfrag[ni] = __builtin_shufflevector(lo, hi, 0,1,2,3,4,5,6,7,8,9,10,11,12,13,14,15);
        }

        #pragma unroll
        for (int mi = 0; mi < 2; ++mi)
            #pragma unroll
            for (int ni = 0; ni < 2; ++ni)
                acc[mi][ni] = __builtin_amdgcn_wmma_f32_16x16x32_f16(
                    false, afrag[mi], false, bfrag[ni],
                    (short)0, acc[mi][ni], false, false);

        __syncthreads();
    }

    // ---- branch-free epilogue ----
    #pragma unroll
    for (int mi = 0; mi < 2; ++mi) {
        #pragma unroll
        for (int ni = 0; ni < 2; ++ni) {
            int n     = bn0 + wn * 32 + ni * 16 + l15;
            int mbase = bm0 + wm * 32 + mi * 16 + ((lane < 16) ? 0 : 8);
            float bval = bias[n];
            #pragma unroll
            for (int r = 0; r < 8; ++r) {
                int m = mbase + r;
                float val = acc[mi][ni][r] + bval;
                if constexpr (RELU)   val = fmaxf(val, 0.0f);
                if constexpr (HASRES) val += Res[(size_t)m * N + n];
                if constexpr (OUTH)   C16[(size_t)m * N + n] = (_Float16)val;
                else                  C32[(size_t)m * N + n] = val;
            }
        }
    }
}

// ---------------- weight transpose-convert: fp32 [K,N] -> f16 [N,K] ----------------
__global__ __launch_bounds__(256)
void transpose_w_kernel(const float* __restrict__ W, _Float16* __restrict__ Wt,
                        int K, int N)
{
    __shared__ float tile[32][33];
    const int k0 = blockIdx.y * 32, n0 = blockIdx.x * 32;
    const int tx = threadIdx.x & 31, ty = threadIdx.x >> 5;   // 32 x 8
    #pragma unroll
    for (int i = 0; i < 4; ++i)
        tile[ty + i * 8][tx] = W[(size_t)(k0 + ty + i * 8) * N + n0 + tx];
    __syncthreads();
    #pragma unroll
    for (int i = 0; i < 4; ++i)
        Wt[(size_t)(n0 + ty + i * 8) * K + k0 + tx] = (_Float16)tile[tx][ty + i * 8];
}

// ---------------- embedding + 2D sinusoidal PE (dual fp32/f16 output) ----------------
__global__ __launch_bounds__(256)
void embed_pe_kernel(const float* __restrict__ x, const float* __restrict__ W_emb,
                     const float* __restrict__ b_emb, float* __restrict__ H,
                     _Float16* __restrict__ H16)
{
    int idx = blockIdx.x * 256 + threadIdx.x;
    int row = idx >> 10;
    int j   = idx & 1023;
    int s   = row & (SEQ - 1);
    float v;
    if (j < 512) {
        v = x[(size_t)row * 2 + 0] * W_emb[j] + x[(size_t)row * 2 + 1] * W_emb[512 + j] + b_emb[j];
    } else {
        int j2 = j - 512;
        int p  = (j2 < 256) ? (s / GRIDW) : (s % GRIDW);
        int ja = (j2 < 256) ? j2 : (j2 - 256);
        float fi  = (float)(2 * (ja >> 1));
        float div = expf(-logf(10000.0f) * fi / 256.0f);
        float ang = (float)p * div;
        v = (ja & 1) ? cosf(ang) : sinf(ang);
    }
    H[idx]   = v;
    H16[idx] = (_Float16)v;
}

// ---------------- chunked causal linear attention (WMMA f32) ----------------
__device__ __forceinline__ float phi_elu1(float x) { return x > 0.0f ? x + 1.0f : expf(x); }

__global__ __launch_bounds__(256)
void attn_chunked(const float* __restrict__ Q, const float* __restrict__ K,
                  const float* __restrict__ V, _Float16* __restrict__ O)
{
    const int b  = blockIdx.x >> 4;
    const int hh = blockIdx.x & 15;

    __shared__ float KV[DHEAD][DHEAD + 4];      // running sum of k^T v (fp32 exact)
    __shared__ float KS[DHEAD];                 // running sum of k
    __shared__ float Qc[CHUNK][DHEAD + 4];
    __shared__ float Kc[CHUNK][DHEAD + 4];
    __shared__ float Vc[CHUNK][DHEAD + 4];
    __shared__ float Ss[CHUNK][CHUNK + 1];      // masked intra-chunk scores
    __shared__ float zbuf[CHUNK];
    __shared__ float zfin[CHUNK];

    const int tid  = threadIdx.x;
    const int lane = tid & 31;
    const int wave = tid >> 5;
    const int l15  = lane & 15;
    const int hi   = (lane < 16) ? 0 : 2;    // K offset for 32-bit A/B fragments
    const int mhi  = (lane < 16) ? 0 : 8;    // C-fragment row offset

    for (int i = tid; i < DHEAD * (DHEAD + 4); i += 256) (&KV[0][0])[i] = 0.0f;
    if (tid < DHEAD) KS[tid] = 0.0f;
    __syncthreads();

    const size_t hoff    = (size_t)hh * DHEAD;
    const size_t rowbase = (size_t)b * SEQ;

    for (int c = 0; c < SEQ / CHUNK; ++c) {
        const int s0 = c * CHUNK;
        // ---- stage chunk: phi(Q), phi(K), V ----
        #pragma unroll
        for (int i = 0; i < 4; ++i) {
            int e = tid + i * 256;
            int r = e >> 6, cc = e & 63;
            size_t g = (rowbase + s0 + r) * DMODEL + hoff + cc;
            Qc[r][cc] = phi_elu1(Q[g]);
            Kc[r][cc] = phi_elu1(K[g]);
            Vc[r][cc] = V[g];
        }
        __syncthreads();

        // ---- phase A: O_inter (waves 0-3), S_intra (wave 4), z_inter (wave 5) ----
        v8f oacc = {};
        if (wave < 4) {
            const int n0 = wave * 16;                   // phiQ @ KV  (K=64)
            #pragma unroll
            for (int kk = 0; kk < 16; ++kk) {
                v2f a, bf;
                a[0]  = Qc[l15][kk * 4 + hi];
                a[1]  = Qc[l15][kk * 4 + hi + 1];
                bf[0] = KV[kk * 4 + hi][n0 + l15];
                bf[1] = KV[kk * 4 + hi + 1][n0 + l15];
                oacc = __builtin_amdgcn_wmma_f32_16x16x4_f32(false, a, false, bf,
                                                             (short)0, oacc, false, false);
            }
        } else if (wave == 4) {                         // S = phiQ @ phiK^T (K=64)
            v8f sacc = {};
            #pragma unroll
            for (int kk = 0; kk < 16; ++kk) {
                v2f a, bf;
                a[0]  = Qc[l15][kk * 4 + hi];
                a[1]  = Qc[l15][kk * 4 + hi + 1];
                bf[0] = Kc[l15][kk * 4 + hi];
                bf[1] = Kc[l15][kk * 4 + hi + 1];
                sacc = __builtin_amdgcn_wmma_f32_16x16x4_f32(false, a, false, bf,
                                                             (short)0, sacc, false, false);
            }
            #pragma unroll
            for (int r = 0; r < 8; ++r) {               // causal mask (inclusive)
                int i = r + mhi;
                Ss[i][l15] = (l15 <= i) ? sacc[r] : 0.0f;
            }
        } else if (wave == 5) {                         // z_inter = phiQ . KS
            if (lane < 16) {
                float z = 0.0f;
                for (int kx = 0; kx < DHEAD; ++kx) z += Qc[lane][kx] * KS[kx];
                zbuf[lane] = z;
            }
        }
        __syncthreads();

        // ---- phase B1: O_intra (waves 0-3), z finalize (wave 4) ----
        if (wave < 4) {
            const int n0 = wave * 16;                   // S_masked @ V  (K=16)
            #pragma unroll
            for (int kk = 0; kk < 4; ++kk) {
                v2f a, bf;
                a[0]  = Ss[l15][kk * 4 + hi];
                a[1]  = Ss[l15][kk * 4 + hi + 1];
                bf[0] = Vc[kk * 4 + hi][n0 + l15];
                bf[1] = Vc[kk * 4 + hi + 1][n0 + l15];
                oacc = __builtin_amdgcn_wmma_f32_16x16x4_f32(false, a, false, bf,
                                                             (short)0, oacc, false, false);
            }
        } else if (wave == 4) {
            if (lane < 16) {
                float zr = zbuf[lane] + 1e-6f;
                #pragma unroll
                for (int j = 0; j < CHUNK; ++j) zr += Ss[lane][j];
                zfin[lane] = zr;
            }
        }
        __syncthreads();

        // ---- phase B2: write output (waves 0-3); update KV,KS (waves 4-7) ----
        if (wave < 4) {
            const int n0 = wave * 16;
            #pragma unroll
            for (int r = 0; r < 8; ++r) {
                int i = r + mhi;
                float val = oacc[r] / zfin[i];
                O[(rowbase + s0 + i) * DMODEL + hoff + n0 + l15] = (_Float16)val;
            }
        } else {
            const int wi = wave - 4;                    // KV += phiK^T @ V  (K=16)
            #pragma unroll
            for (int tix = 0; tix < 4; ++tix) {
                const int ti = wi * 4 + tix;
                const int m0 = (ti >> 2) * 16;
                const int n0 = (ti & 3) * 16;
                v8f acc;
                #pragma unroll
                for (int r = 0; r < 8; ++r) acc[r] = KV[m0 + r + mhi][n0 + l15];
                #pragma unroll
                for (int kk = 0; kk < 4; ++kk) {
                    v2f a, bf;
                    a[0]  = Kc[kk * 4 + hi][m0 + l15];
                    a[1]  = Kc[kk * 4 + hi + 1][m0 + l15];
                    bf[0] = Vc[kk * 4 + hi][n0 + l15];
                    bf[1] = Vc[kk * 4 + hi + 1][n0 + l15];
                    acc = __builtin_amdgcn_wmma_f32_16x16x4_f32(false, a, false, bf,
                                                                (short)0, acc, false, false);
                }
                #pragma unroll
                for (int r = 0; r < 8; ++r) KV[m0 + r + mhi][n0 + l15] = acc[r];
            }
            if (wave == 5 || wave == 6) {               // KS += colsum(phiK)
                int n = (wave - 5) * 32 + lane;
                float sacc = 0.0f;
                #pragma unroll
                for (int i = 0; i < CHUNK; ++i) sacc += Kc[i][n];
                KS[n] += sacc;
            }
        }
        __syncthreads();
    }
}

// ---------------- LayerNorm (dual fp32/f16 output) ----------------
__global__ __launch_bounds__(256)
void ln_kernel(const float* __restrict__ X, const float* __restrict__ g,
               const float* __restrict__ bb, float* __restrict__ Y,
               _Float16* __restrict__ Y16)
{
    const int row = blockIdx.x;
    const int tid = threadIdx.x;
    const float* x = X + (size_t)row * DMODEL;
    __shared__ float red[256];

    float4 v = *(const float4*)(x + tid * 4);
    red[tid] = v.x + v.y + v.z + v.w;
    __syncthreads();
    for (int off = 128; off > 0; off >>= 1) { if (tid < off) red[tid] += red[tid + off]; __syncthreads(); }
    float mean = red[0] * (1.0f / DMODEL);
    __syncthreads();

    float dx = v.x - mean, dy = v.y - mean, dz = v.z - mean, dw = v.w - mean;
    red[tid] = dx * dx + dy * dy + dz * dz + dw * dw;
    __syncthreads();
    for (int off = 128; off > 0; off >>= 1) { if (tid < off) red[tid] += red[tid + off]; __syncthreads(); }
    float rstd = rsqrtf(red[0] * (1.0f / DMODEL) + 1e-5f);

    int c = tid * 4;
    float o0 = dx * rstd * g[c + 0] + bb[c + 0];
    float o1 = dy * rstd * g[c + 1] + bb[c + 1];
    float o2 = dz * rstd * g[c + 2] + bb[c + 2];
    float o3 = dw * rstd * g[c + 3] + bb[c + 3];
    float* y = Y + (size_t)row * DMODEL;
    y[c + 0] = o0; y[c + 1] = o1; y[c + 2] = o2; y[c + 3] = o3;
    _Float16* y16 = Y16 + (size_t)row * DMODEL;
    y16[c + 0] = (_Float16)o0; y16[c + 1] = (_Float16)o1;
    y16[c + 2] = (_Float16)o2; y16[c + 3] = (_Float16)o3;
}

// ---------------- output head ----------------
__global__ __launch_bounds__(256)
void head_kernel(const float* __restrict__ H, const float* __restrict__ Wa,
                 const float* __restrict__ ba, const float* __restrict__ Wp,
                 const float* __restrict__ bp, float* __restrict__ Out)
{
    const int row = blockIdx.x;
    const int tid = threadIdx.x;
    const float* h = H + (size_t)row * DMODEL;
    __shared__ float ra[256], rp[256];

    float4 hv = *(const float4*)(h + tid * 4);
    float4 wa = *(const float4*)(Wa + tid * 4);
    float4 wp = *(const float4*)(Wp + tid * 4);
    ra[tid] = hv.x * wa.x + hv.y * wa.y + hv.z * wa.z + hv.w * wa.w;
    rp[tid] = hv.x * wp.x + hv.y * wp.y + hv.z * wp.z + hv.w * wp.w;
    __syncthreads();
    for (int off = 128; off > 0; off >>= 1) {
        if (tid < off) { ra[tid] += ra[tid + off]; rp[tid] += rp[tid + off]; }
        __syncthreads();
    }
    if (tid == 0) {
        Out[(size_t)row * 2 + 0] = ra[0] + ba[0];
        Out[(size_t)row * 2 + 1] = tanhf(rp[0] + bp[0]);
    }
}

// ---------------- host orchestration ----------------
extern "C" void kernel_launch(void* const* d_in, const int* in_sizes, int n_in,
                              void* d_out, int out_size, void* d_ws, size_t ws_size,
                              hipStream_t stream)
{
    const float* x     = (const float*)d_in[0];
    const float* W_emb = (const float*)d_in[1];
    const float* b_emb = (const float*)d_in[2];
    const float* Wq    = (const float*)d_in[3];
    const float* bq    = (const float*)d_in[4];
    const float* Wk    = (const float*)d_in[5];
    const float* bk    = (const float*)d_in[6];
    const float* Wv    = (const float*)d_in[7];
    const float* bv    = (const float*)d_in[8];
    const float* Wo    = (const float*)d_in[9];
    const float* bo    = (const float*)d_in[10];
    const float* g1    = (const float*)d_in[11];
    const float* b1    = (const float*)d_in[12];
    const float* W1    = (const float*)d_in[13];
    const float* bf1   = (const float*)d_in[14];
    const float* W2    = (const float*)d_in[15];
    const float* bf2   = (const float*)d_in[16];
    const float* g2    = (const float*)d_in[17];
    const float* b2    = (const float*)d_in[18];
    const float* Wa    = (const float*)d_in[19];
    const float* ba    = (const float*)d_in[20];
    const float* Wp    = (const float*)d_in[21];
    const float* bp    = (const float*)d_in[22];
    float* out = (float*)d_out;

    const size_t ND = (size_t)MROWS * DMODEL;       // 4,194,304
    const size_t DD = (size_t)DMODEL * DMODEL;      // 1,048,576
    float* ws = (float*)d_ws;
    float* h  = ws;             // fp32 activations
    float* q  = ws + 1 * ND;
    float* k  = ws + 2 * ND;
    float* v  = ws + 3 * ND;
    float* t2 = ws + 4 * ND;

    _Float16* hp = (_Float16*)(ws + 5 * ND);        // f16 region
    _Float16* h16 = hp;                             // [MROWS, DMODEL]
    _Float16* a16 = hp + 1 * ND;                    // attention out
    _Float16* t1h = hp + 2 * ND;                    // [MROWS, DFF]
    _Float16* wqt = hp + 6 * ND;                    // transposed f16 weights
    _Float16* wkt = wqt + DD;
    _Float16* wvt = wkt + DD;
    _Float16* wot = wvt + DD;
    _Float16* w1t = wot + DD;                       // [DFF, DMODEL]
    _Float16* w2t = w1t + 4 * DD;                   // [DMODEL, DFF]

    const dim3 blk(256);
    const dim3 gEmb((MROWS * (size_t)DMODEL) / 256);
    const dim3 gProj(DMODEL / BN, MROWS / BM);      // (16, 32)
    const dim3 gFfn1(DFF / BN,    MROWS / BM);      // (64, 32)
    const dim3 gRow(MROWS);
    const dim3 gAttn(BATCH * NHEAD);
    const dim3 gTrDD(DMODEL / 32, DMODEL / 32);     // (32, 32)
    const dim3 gTrW1(DFF / 32,    DMODEL / 32);     // (128, 32)
    const dim3 gTrW2(DMODEL / 32, DFF / 32);        // (32, 128)

    embed_pe_kernel<<<gEmb, blk, 0, stream>>>(x, W_emb, b_emb, h, h16);

    for (int l = 0; l < NLAYER; ++l) {
        const size_t wOff  = (size_t)l * DD;
        const size_t bOff  = (size_t)l * DMODEL;
        const size_t fOff  = (size_t)l * DMODEL * DFF;
        const size_t fbOff = (size_t)l * DFF;

        transpose_w_kernel<<<gTrDD, blk, 0, stream>>>(Wq + wOff, wqt, DMODEL, DMODEL);
        transpose_w_kernel<<<gTrDD, blk, 0, stream>>>(Wk + wOff, wkt, DMODEL, DMODEL);
        transpose_w_kernel<<<gTrDD, blk, 0, stream>>>(Wv + wOff, wvt, DMODEL, DMODEL);
        transpose_w_kernel<<<gTrDD, blk, 0, stream>>>(Wo + wOff, wot, DMODEL, DMODEL);
        transpose_w_kernel<<<gTrW1, blk, 0, stream>>>(W1 + fOff, w1t, DMODEL, DFF);
        transpose_w_kernel<<<gTrW2, blk, 0, stream>>>(W2 + fOff, w2t, DFF, DMODEL);

        gemm_wmma<false,false,false><<<gProj, blk, 0, stream>>>(
            h16, wqt, bq + bOff, nullptr, q, nullptr, MROWS, DMODEL, DMODEL);
        gemm_wmma<false,false,false><<<gProj, blk, 0, stream>>>(
            h16, wkt, bk + bOff, nullptr, k, nullptr, MROWS, DMODEL, DMODEL);
        gemm_wmma<false,false,false><<<gProj, blk, 0, stream>>>(
            h16, wvt, bv + bOff, nullptr, v, nullptr, MROWS, DMODEL, DMODEL);

        attn_chunked<<<gAttn, blk, 0, stream>>>(q, k, v, a16);

        gemm_wmma<false,true,false><<<gProj, blk, 0, stream>>>(
            a16, wot, bo + bOff, h, t2, nullptr, MROWS, DMODEL, DMODEL);
        ln_kernel<<<gRow, blk, 0, stream>>>(t2, g1 + bOff, b1 + bOff, h, h16);

        gemm_wmma<true,false,true><<<gFfn1, blk, 0, stream>>>(
            h16, w1t, bf1 + fbOff, nullptr, nullptr, t1h, MROWS, DFF, DMODEL);
        gemm_wmma<false,true,false><<<gProj, blk, 0, stream>>>(
            t1h, w2t, bf2 + bOff, h, t2, nullptr, MROWS, DMODEL, DFF);
        ln_kernel<<<gRow, blk, 0, stream>>>(t2, g2 + bOff, b2 + bOff, h, h16);
    }

    head_kernel<<<gRow, blk, 0, stream>>>(h, Wa, ba, Wp, bp, out);
}
